// WeakTieNet_22488448762503
// MI455X (gfx1250) — compile-verified, matrix operand order checked by reading, not run on hardware
//
#include <hip/hip_runtime.h>
#include <hip/hip_bf16.h>
#include <cstddef>

// ---------------------------------------------------------------------------
// Shapes: B=8, N=256, F=128, H=128, O=64.  Rows = B*N = 2048.
// Pipeline (all bf16-in / f32-acc WMMA for the GEMMs, 32x32 macro-tiles):
//   prep  : bf16 conversions, global_x transpose, combined local/key slices,
//           denom row-sums
//   k_weak: weak = (mask @ global_x)/denom  -> combined[:,128:256] (bf16)
//   k_fc  : h = combined @ fc_w^T + fc_b    (f32)
//   k_ln  : LayerNorm + ReLU -> x (bf16)
//   k_gru : gi = x @ w_ih^T + b_ih ; gh = hprev @ w_hh^T + b_hh (f32)
//   k_gate: GRU gates -> h_new (f32 to d_out) + h_new bf16
//   k_out : out = h_new @ out_w^T + out_b  (f32 to d_out)
// ---------------------------------------------------------------------------

typedef __attribute__((ext_vector_type(16))) __bf16 v16bf;
typedef __attribute__((ext_vector_type(8)))  __bf16 v8bf;
typedef __attribute__((ext_vector_type(8)))  float  v8f;

#define BB 8
#define NN 256
#define FF 128
#define HH 128
#define OO 64
#define ROWS (BB * NN)      // 2048
#define C3F (3 * FF)        // 384
#define H3 (3 * HH)         // 384

// ---- workspace layout (bytes). Regions overlap where lifetimes allow. ----
// Q1 [0, 3145728): phase A = maskb | gxTb | combinedb ; phase B = gi
#define OFF_MASKB   ((size_t)0)          // bf16 [8][256][256]  = 1048576 B
#define OFF_GXT     ((size_t)1048576)    // bf16 [8][128][256]  =  524288 B
#define OFF_COMB    ((size_t)1572864)    // bf16 [2048][384]    = 1572864 B
#define OFF_GI      ((size_t)0)          // f32  [2048][384]    = 3145728 B
// Q2 [3145728, 6291456): phase A = h (f32, 1 MB) ; phase B = gh
#define OFF_H       ((size_t)3145728)    // f32  [2048][128]    = 1048576 B
#define OFF_GH      ((size_t)3145728)    // f32  [2048][384]    = 3145728 B
// P  [6291456, ...): persistent small buffers
#define OFF_FCW     ((size_t)6291456)    // bf16 [128][384]
#define OFF_WIH     ((size_t)6389760)    // bf16 [384][128]
#define OFF_WHH     ((size_t)6488064)    // bf16 [384][128]
#define OFF_OUTW    ((size_t)6586368)    // bf16 [64][128]
#define OFF_HPREVB  ((size_t)6602752)    // bf16 [2048][128]
#define OFF_DENOM   ((size_t)7127040)    // f32  [2048]
#define OFF_XB      ((size_t)7135232)    // bf16 [2048][128]
#define OFF_HNEWB   ((size_t)7659520)    // bf16 [2048][128]
// total = 8183808 bytes (~7.8 MB)

// ---------------------------------------------------------------------------
// Core: one wave computes a 32x32 f32 macro-tile = 2x2 WMMA tiles, K steps
// of 32 (bf16 WMMA, f32 accumulate).  Each A/B fragment is reused twice:
// 8 x 16B loads feed 4 v_wmma per K-step (32 B per WMMA vs 64 B unblocked).
// A fragment: lane l -> row (l&15), k-base (l>>4)*8; elems 0..7 = k..k+7,
// elems 8..15 = k+16..k+23.  B mirrors with column (l&15).
// C/D: lane l holds N=(l&15); VGPR v holds M = v + 8*(l>>4).
// ---------------------------------------------------------------------------
__device__ __forceinline__ v16bf load_frag(const __bf16* p) {
    union { v16bf v; v8bf h[2]; } f;
    f.h[0] = *(const v8bf*)(p);
    f.h[1] = *(const v8bf*)(p + 16);
    return f.v;
}

__device__ __forceinline__ void wmma_tile_2x2(const __bf16* __restrict__ A, int lda,
                                              const __bf16* __restrict__ B, int ldb,
                                              int K, v8f acc[4]) {
    const int lane = threadIdx.x & 31;
    const int half = lane >> 4;
    const int rc   = lane & 15;
    const __bf16* a0 = A + (size_t)rc * lda;            // rows 0..15
    const __bf16* a1 = a0 + (size_t)16 * lda;           // rows 16..31
    const __bf16* b0 = B + (size_t)rc * ldb;            // cols 0..15
    const __bf16* b1 = b0 + (size_t)16 * ldb;           // cols 16..31
    for (int k0 = 0; k0 < K; k0 += 32) {
        const int kb = k0 + half * 8;
        v16bf fa0 = load_frag(a0 + kb);
        v16bf fa1 = load_frag(a1 + kb);
        v16bf fb0 = load_frag(b0 + kb);
        v16bf fb1 = load_frag(b1 + kb);
        acc[0] = __builtin_amdgcn_wmma_f32_16x16x32_bf16(false, fa0, false, fb0, (short)0, acc[0], false, false);
        acc[1] = __builtin_amdgcn_wmma_f32_16x16x32_bf16(false, fa0, false, fb1, (short)0, acc[1], false, false);
        acc[2] = __builtin_amdgcn_wmma_f32_16x16x32_bf16(false, fa1, false, fb0, (short)0, acc[2], false, false);
        acc[3] = __builtin_amdgcn_wmma_f32_16x16x32_bf16(false, fa1, false, fb1, (short)0, acc[3], false, false);
    }
}

#define ACC_INIT(a) v8f a[4]; \
    a[0] = (v8f){0.f,0.f,0.f,0.f,0.f,0.f,0.f,0.f}; \
    a[1] = a[0]; a[2] = a[0]; a[3] = a[0];

// ------------------------------- prep kernels ------------------------------

__global__ void k_cvt_bf16(__bf16* __restrict__ dst, const float* __restrict__ src, int n) {
    int i = blockIdx.x * blockDim.x + threadIdx.x;
    if (i < n) dst[i] = (__bf16)src[i];
}

// gxT[b][f][k] = global_x[b][k][f]   (k = agent index, contiguous in gxT)
__global__ void k_transpose_gx(__bf16* __restrict__ gxT, const float* __restrict__ gx) {
    int i = blockIdx.x * blockDim.x + threadIdx.x;   // < 8*256*128
    if (i >= BB * NN * FF) return;
    int b = i >> 15;            // /32768
    int r = i & 32767;
    int k = r >> 7;             // agent 0..255
    int f = r & 127;            // feature 0..127
    gxT[((size_t)b * FF + f) * NN + k] = (__bf16)gx[i];
}

// combined[:,0:128] = local_x ; combined[:,256:384] = key_feat
__global__ void k_fill_combined(__bf16* __restrict__ comb,
                                const float* __restrict__ local_x,
                                const float* __restrict__ gx,
                                const int* __restrict__ key_idx) {
    int i = blockIdx.x * blockDim.x + threadIdx.x;   // < 2048*128
    if (i >= ROWS * FF) return;
    int row = i >> 7;
    int c   = i & 127;
    int b   = row >> 8;
    comb[(size_t)row * C3F + c] = (__bf16)local_x[i];
    float kf = gx[((size_t)b * NN + key_idx[b]) * FF + c];
    comb[(size_t)row * C3F + 2 * FF + c] = (__bf16)kf;
}

// denom[row] = sum_j mask[row][j] + 1e-6
__global__ void k_denom(float* __restrict__ denom, const float* __restrict__ mask) {
    int row = blockIdx.x * blockDim.x + threadIdx.x;
    if (row >= ROWS) return;
    const float* m = mask + (size_t)row * NN;
    float s = 0.f;
    for (int j = 0; j < NN; ++j) s += m[j];
    denom[row] = s + 1e-6f;
}

// ------------------------------- WMMA GEMMs --------------------------------

// weak = (mask @ global_x) / denom  -> combined[:,128:256] (bf16)
// waves: b(8) x mt(8 of 32 rows) x nt(4 of 32 cols) = 256
__global__ __launch_bounds__(128) void k_weak(const __bf16* __restrict__ maskb,
                                              const __bf16* __restrict__ gxT,
                                              const float* __restrict__ denom,
                                              __bf16* __restrict__ comb) {
    int wave = blockIdx.x * (blockDim.x >> 5) + (threadIdx.x >> 5);
    int nt = wave & 3;
    int mt = (wave >> 2) & 7;
    int b  = wave >> 5;
    const __bf16* A = maskb + ((size_t)b * NN + mt * 32) * NN;
    const __bf16* B = gxT   + ((size_t)b * FF + nt * 32) * NN;
    ACC_INIT(acc);
    wmma_tile_2x2(A, NN, B, NN, NN, acc);
    const int lane = threadIdx.x & 31, half = lane >> 4, rc = lane & 15;
#pragma unroll
    for (int mi = 0; mi < 2; ++mi)
#pragma unroll
        for (int ni = 0; ni < 2; ++ni) {
            const v8f& a = acc[mi * 2 + ni];
            int n = nt * 32 + ni * 16 + rc;
#pragma unroll
            for (int v = 0; v < 8; ++v) {
                int m   = mt * 32 + mi * 16 + half * 8 + v;
                int row = b * NN + m;
                float val = a[v] / denom[row];
                comb[(size_t)row * C3F + FF + n] = (__bf16)val;
            }
        }
}

// h = combined @ fc_w^T + fc_b   waves: mt(64) x nt(4) = 256
__global__ __launch_bounds__(128) void k_fc(const __bf16* __restrict__ comb,
                                            const __bf16* __restrict__ fcw,
                                            const float* __restrict__ fcb,
                                            float* __restrict__ h) {
    int wave = blockIdx.x * (blockDim.x >> 5) + (threadIdx.x >> 5);
    int nt = wave & 3;
    int mt = wave >> 2;
    const __bf16* A = comb + (size_t)mt * 32 * C3F;
    const __bf16* B = fcw  + (size_t)nt * 32 * C3F;
    ACC_INIT(acc);
    wmma_tile_2x2(A, C3F, B, C3F, C3F, acc);
    const int lane = threadIdx.x & 31, half = lane >> 4, rc = lane & 15;
#pragma unroll
    for (int mi = 0; mi < 2; ++mi)
#pragma unroll
        for (int ni = 0; ni < 2; ++ni) {
            const v8f& a = acc[mi * 2 + ni];
            int n = nt * 32 + ni * 16 + rc;
            float bias = fcb[n];
#pragma unroll
            for (int v = 0; v < 8; ++v) {
                int m = mt * 32 + mi * 16 + half * 8 + v;
                h[(size_t)m * HH + n] = a[v] + bias;
            }
        }
}

// LayerNorm + ReLU, one wave per row (wave32), -> x bf16
__global__ __launch_bounds__(256) void k_ln(const float* __restrict__ h,
                                            const float* __restrict__ g,
                                            const float* __restrict__ bta,
                                            __bf16* __restrict__ x) {
    int row  = blockIdx.x * (blockDim.x >> 5) + (threadIdx.x >> 5);   // 2048 rows
    int lane = threadIdx.x & 31;
    const float* hr = h + (size_t)row * HH;
    float vals[4], s = 0.f, s2 = 0.f;
#pragma unroll
    for (int i = 0; i < 4; ++i) {
        float v = hr[lane + i * 32];
        vals[i] = v; s += v; s2 += v * v;
    }
#pragma unroll
    for (int off = 16; off > 0; off >>= 1) {
        s  += __shfl_xor(s,  off, 32);
        s2 += __shfl_xor(s2, off, 32);
    }
    float mu  = s * (1.f / HH);
    float var = s2 * (1.f / HH) - mu * mu;
    float inv = rsqrtf(var + 1e-5f);
#pragma unroll
    for (int i = 0; i < 4; ++i) {
        int c = lane + i * 32;
        float y = (vals[i] - mu) * inv * g[c] + bta[c];
        y = fmaxf(y, 0.f);
        x[(size_t)row * HH + c] = (__bf16)y;
    }
}

// gi = x @ w_ih^T + b_ih ; gh = hprev @ w_hh^T + b_hh
// per-GEMM waves: mt(64) x nt(12) = 768 ; total 1536
__global__ __launch_bounds__(128) void k_gru_gemm(const __bf16* __restrict__ x,
                                                  const __bf16* __restrict__ hprevb,
                                                  const __bf16* __restrict__ wih,
                                                  const __bf16* __restrict__ whh,
                                                  const float* __restrict__ bih,
                                                  const float* __restrict__ bhh,
                                                  float* __restrict__ gi,
                                                  float* __restrict__ gh) {
    int wave = blockIdx.x * (blockDim.x >> 5) + (threadIdx.x >> 5);
    const int per = (ROWS / 32) * (H3 / 32);   // 64 * 12 = 768
    int which = wave >= per;
    int t  = which ? wave - per : wave;
    int nt = t % (H3 / 32);
    int mt = t / (H3 / 32);
    const __bf16* A = (which ? hprevb : x)   + (size_t)mt * 32 * HH;
    const __bf16* B = (which ? whh    : wih) + (size_t)nt * 32 * HH;
    const float* bias = which ? bhh : bih;
    float* out = which ? gh : gi;
    ACC_INIT(acc);
    wmma_tile_2x2(A, HH, B, HH, HH, acc);
    const int lane = threadIdx.x & 31, half = lane >> 4, rc = lane & 15;
#pragma unroll
    for (int mi = 0; mi < 2; ++mi)
#pragma unroll
        for (int ni = 0; ni < 2; ++ni) {
            const v8f& a = acc[mi * 2 + ni];
            int n = nt * 32 + ni * 16 + rc;
            float bv = bias[n];
#pragma unroll
            for (int v = 0; v < 8; ++v) {
                int m = mt * 32 + mi * 16 + half * 8 + v;
                out[(size_t)m * H3 + n] = a[v] + bv;
            }
        }
}

// GRU gate math -> h_new (f32 to d_out) and h_new bf16
__global__ void k_gate(const float* __restrict__ gi, const float* __restrict__ gh,
                       const float* __restrict__ hprev_f,
                       float* __restrict__ hnew_out, __bf16* __restrict__ hnewb) {
    int i = blockIdx.x * blockDim.x + threadIdx.x;   // < 2048*128
    if (i >= ROWS * HH) return;
    int row = i >> 7;
    int c   = i & 127;
    const float* gir = gi + (size_t)row * H3;
    const float* ghr = gh + (size_t)row * H3;
    float r  = 1.f / (1.f + __expf(-(gir[c]          + ghr[c])));
    float z  = 1.f / (1.f + __expf(-(gir[HH + c]     + ghr[HH + c])));
    float nn = tanhf(gir[2 * HH + c] + r * ghr[2 * HH + c]);
    float hf = hprev_f[i];
    float hn = (1.f - z) * nn + z * hf;
    hnew_out[i] = hn;
    hnewb[i]    = (__bf16)hn;
}

// out = h_new @ out_w^T + out_b   waves: mt(64) x nt(2) = 128
__global__ __launch_bounds__(128) void k_out(const __bf16* __restrict__ hnewb,
                                             const __bf16* __restrict__ outw,
                                             const float* __restrict__ outb,
                                             float* __restrict__ out) {
    int wave = blockIdx.x * (blockDim.x >> 5) + (threadIdx.x >> 5);
    int nt = wave & 1;
    int mt = wave >> 1;
    const __bf16* A = hnewb + (size_t)mt * 32 * HH;
    const __bf16* B = outw  + (size_t)nt * 32 * HH;
    ACC_INIT(acc);
    wmma_tile_2x2(A, HH, B, HH, HH, acc);
    const int lane = threadIdx.x & 31, half = lane >> 4, rc = lane & 15;
#pragma unroll
    for (int mi = 0; mi < 2; ++mi)
#pragma unroll
        for (int ni = 0; ni < 2; ++ni) {
            const v8f& a = acc[mi * 2 + ni];
            int n = nt * 32 + ni * 16 + rc;
            float bias = outb[n];
#pragma unroll
            for (int v = 0; v < 8; ++v) {
                int m = mt * 32 + mi * 16 + half * 8 + v;
                out[(size_t)m * OO + n] = a[v] + bias;
            }
        }
}

// ---------------------------------------------------------------------------

extern "C" void kernel_launch(void* const* d_in, const int* in_sizes, int n_in,
                              void* d_out, int out_size, void* d_ws, size_t ws_size,
                              hipStream_t stream) {
    const float* local_x  = (const float*)d_in[0];
    const float* global_x = (const float*)d_in[1];
    const float* mask     = (const float*)d_in[2];
    const int*   key_idx  = (const int*)d_in[3];
    const float* hidden   = (const float*)d_in[4];
    const float* fc_w     = (const float*)d_in[5];
    const float* fc_b     = (const float*)d_in[6];
    const float* ln_g     = (const float*)d_in[7];
    const float* ln_b     = (const float*)d_in[8];
    const float* w_ih     = (const float*)d_in[9];
    const float* w_hh     = (const float*)d_in[10];
    const float* b_ih     = (const float*)d_in[11];
    const float* b_hh     = (const float*)d_in[12];
    const float* out_w    = (const float*)d_in[13];
    const float* out_b    = (const float*)d_in[14];

    char* ws = (char*)d_ws;
    __bf16* maskb  = (__bf16*)(ws + OFF_MASKB);
    __bf16* gxT    = (__bf16*)(ws + OFF_GXT);
    __bf16* comb   = (__bf16*)(ws + OFF_COMB);
    float*  gi     = (float*) (ws + OFF_GI);
    float*  hbuf   = (float*) (ws + OFF_H);
    float*  gh     = (float*) (ws + OFF_GH);
    __bf16* fcwb   = (__bf16*)(ws + OFF_FCW);
    __bf16* wihb   = (__bf16*)(ws + OFF_WIH);
    __bf16* whhb   = (__bf16*)(ws + OFF_WHH);
    __bf16* outwb  = (__bf16*)(ws + OFF_OUTW);
    __bf16* hprevb = (__bf16*)(ws + OFF_HPREVB);
    float*  denom  = (float*) (ws + OFF_DENOM);
    __bf16* xb     = (__bf16*)(ws + OFF_XB);
    __bf16* hnewb  = (__bf16*)(ws + OFF_HNEWB);

    float* out_part  = (float*)d_out;              // [2048][64]
    float* hnew_part = out_part + ROWS * OO;       // [2048][128]

    // ---- prep: bf16 conversions / layouts ----
    k_cvt_bf16<<<(BB*NN*NN + 255) / 256, 256, 0, stream>>>(maskb,  mask,   BB*NN*NN);
    k_cvt_bf16<<<(HH*C3F   + 255) / 256, 256, 0, stream>>>(fcwb,   fc_w,   HH*C3F);
    k_cvt_bf16<<<(H3*HH    + 255) / 256, 256, 0, stream>>>(wihb,   w_ih,   H3*HH);
    k_cvt_bf16<<<(H3*HH    + 255) / 256, 256, 0, stream>>>(whhb,   w_hh,   H3*HH);
    k_cvt_bf16<<<(OO*HH    + 255) / 256, 256, 0, stream>>>(outwb,  out_w,  OO*HH);
    k_cvt_bf16<<<(ROWS*HH  + 255) / 256, 256, 0, stream>>>(hprevb, hidden, ROWS*HH);
    k_transpose_gx<<<(BB*NN*FF + 255) / 256, 256, 0, stream>>>(gxT, global_x);
    k_fill_combined<<<(ROWS*FF + 255) / 256, 256, 0, stream>>>(comb, local_x, global_x, key_idx);
    k_denom<<<(ROWS + 255) / 256, 256, 0, stream>>>(denom, mask);

    // ---- WMMA pipeline (32x32 macro-tiles per wave) ----
    k_weak<<<64, 128, 0, stream>>>(maskb, gxT, denom, comb);           // 256 waves
    k_fc<<<64, 128, 0, stream>>>(comb, fcwb, fc_b, hbuf);              // 256 waves
    k_ln<<<256, 256, 0, stream>>>(hbuf, ln_g, ln_b, xb);               // 2048 rows
    k_gru_gemm<<<384, 128, 0, stream>>>(xb, hprevb, wihb, whhb,
                                        b_ih, b_hh, gi, gh);           // 1536 waves
    k_gate<<<(ROWS*HH + 255) / 256, 256, 0, stream>>>(gi, gh, hidden,
                                                      hnew_part, hnewb);
    k_out<<<32, 128, 0, stream>>>(hnewb, outwb, out_b, out_part);      // 128 waves
}
